// DRAL_88476326297980
// MI455X (gfx1250) — compile-verified, matrix operand order checked by reading, not run on hardware
//
#include <hip/hip_runtime.h>
#include <hip/hip_bf16.h>

typedef __bf16 bf16_t;
typedef __attribute__((ext_vector_type(16))) __bf16 v16bf;
typedef __attribute__((ext_vector_type(8)))  __bf16 v8bf;
typedef __attribute__((ext_vector_type(4)))  __bf16 v4bf;
typedef __attribute__((ext_vector_type(2)))  __bf16 v2bf;
typedef __attribute__((ext_vector_type(8)))  float  v8f;
typedef __attribute__((ext_vector_type(4)))  float  v4f;

// Flip to 0 to fall back to register-staged LDS fills.
#define USE_ASYNC_LDS 1

// ---------------------------------------------------------------------------
// CDNA5 async global->LDS copy (ISA §10 / §15.18, opcode 98), ASYNCcnt-tracked.
// Asm operands per VGLOBAL field table: VDST = per-lane LDS byte address,
// VADDR = 64-bit global address, no SADDR ("off").
// ---------------------------------------------------------------------------
__device__ inline void async_ld16(bf16_t* l, const bf16_t* g) {
    unsigned laddr = (unsigned)(unsigned long long)l;   // LDS aperture: low 32b
    unsigned long long ga = (unsigned long long)g;
    asm volatile("global_load_async_to_lds_b128 %0, %1, off"
                 :: "v"(laddr), "v"(ga) : "memory");
}

// ---------------------------------------------------------------------------
// WMMA fragment load (CDNA5 wave32, 16-bit A/B 16x32 layout, ISA 7.12.2):
// lane L<16 holds row M=L with K={0..7,16..23}; lane L>=16 holds row M=L-16
// with K={8..15,24..31}.  Tiles stored [row][k] row-major in LDS; two
// ds_load_b128 per fragment.
// ---------------------------------------------------------------------------
__device__ inline v16bf load_frag(const bf16_t* __restrict__ base,
                                  int row0, int k0, int ld) {
    const int lane = threadIdx.x & 31;
    const int r    = row0 + (lane & 15);
    const int kh   = (lane >> 4) & 1;
    const bf16_t* p = base + (size_t)r * ld + k0 + kh * 8;
    v8bf lo = *(const v8bf*)(p);
    v8bf hi = *(const v8bf*)(p + 16);
    return __builtin_shufflevector(lo, hi,
        0,1,2,3,4,5,6,7,8,9,10,11,12,13,14,15);
}

__device__ inline v8f wmma_bf16(v16bf a, v16bf b, v8f c) {
    return __builtin_amdgcn_wmma_f32_16x16x32_bf16(
        false, a, false, b, (short)0, c, false, false);
}

// ---------------------------------------------------------------------------
// Kernel 1: 1x1 conv as GEMM  O(128 x 4096) = W(128x256) @ X(256x4096) + bias,
// bf16 out.  mode 0: out[b][n][128] (theta/phi)   mode 1: out[b][ci][4096] (g)
// Block 256 thr (8 waves); tile 128ci x 64n; K stepped by 32, double-buffered.
// ---------------------------------------------------------------------------
__global__ __launch_bounds__(256)
void proj_kernel(const float* __restrict__ X,    // (B,256,4096)
                 const float* __restrict__ Wm,   // (128,256)
                 const float* __restrict__ bias, // (128)
                 bf16_t* __restrict__ out,
                 int mode) {
    const int C = 256, N = 4096;
    __shared__ __align__(16) bf16_t sA[2][128 * 32];  // [m=ci][k]
    __shared__ __align__(16) bf16_t sB[2][64 * 32];   // [n][k]
    __shared__ __align__(16) bf16_t sOut[128 * 64];   // staged epilogue

    const int b     = blockIdx.z;
    const int nBase = blockIdx.x * 64;
    const int tid = threadIdx.x, wave = tid >> 5, lane = tid & 31;
    const int lncol = lane & 15, rowh = (lane >> 4) * 8;

    const int am = tid >> 1, ah = (tid & 1) * 16;     // A: row, k-offset
    const int kp = tid >> 4, n4 = (tid & 15) * 4;     // B: k-pair, n-quad

    const float* Arow = Wm + (size_t)am * C + ah;
    const float* Bcol = X + (size_t)b * C * N + nBase + n4;

    v4f ra[4], rb0, rb1;
    auto loadStage = [&](int k0) {
#pragma unroll
        for (int q = 0; q < 4; ++q) ra[q] = *(const v4f*)(Arow + k0 + q * 4);
        rb0 = *(const v4f*)(Bcol + (size_t)(k0 + 2 * kp) * N);
        rb1 = *(const v4f*)(Bcol + (size_t)(k0 + 2 * kp + 1) * N);
    };
    auto storeStage = [&](int buf) {
        v8bf pa0, pa1;
#pragma unroll
        for (int j = 0; j < 8; ++j) pa0[j] = (bf16_t)ra[j >> 2][j & 3];
#pragma unroll
        for (int j = 0; j < 8; ++j) pa1[j] = (bf16_t)ra[2 + (j >> 2)][j & 3];
        *(v8bf*)&sA[buf][am * 32 + ah]     = pa0;
        *(v8bf*)&sA[buf][am * 32 + ah + 8] = pa1;
#pragma unroll
        for (int j = 0; j < 4; ++j) {
            v2bf pk; pk[0] = (bf16_t)rb0[j]; pk[1] = (bf16_t)rb1[j];
            *(v2bf*)&sB[buf][(n4 + j) * 32 + 2 * kp] = pk;   // packed k-pair
        }
    };

    v8f acc[4] = {};
    loadStage(0);
    storeStage(0);
    __syncthreads();
    for (int s = 0; s < 8; ++s) {
        if (s < 7) loadStage((s + 1) * 32);          // overlap with WMMA
        v16bf a = load_frag(sA[s & 1], wave * 16, 0, 32);
#pragma unroll
        for (int nt = 0; nt < 4; ++nt) {
            v16bf bb = load_frag(sB[s & 1], nt * 16, 0, 32);
            acc[nt] = wmma_bf16(a, bb, acc[nt]);
        }
        if (s < 7) storeStage((s + 1) & 1);
        __syncthreads();
    }

    // epilogue: bias, stage transposed tile in LDS, write coalesced b128 runs
#pragma unroll
    for (int nt = 0; nt < 4; ++nt)
#pragma unroll
        for (int r = 0; r < 8; ++r) {
            int ci = wave * 16 + r + rowh;
            int nl = nt * 16 + lncol;
            float v = acc[nt][r] + bias[ci];
            if (mode == 0) sOut[nl * 128 + ci] = (bf16_t)v;
            else           sOut[ci * 64 + nl] = (bf16_t)v;
        }
    __syncthreads();
    if (mode == 0) {
        bf16_t* dst = out + ((size_t)b * 4096 + nBase) * 128; // contiguous 16KB
#pragma unroll
        for (int c = 0; c < 4; ++c) {
            int idx = tid + c * 256;
            *(v8bf*)(dst + idx * 8) = *(const v8bf*)(sOut + idx * 8);
        }
    } else {
#pragma unroll
        for (int c = 0; c < 4; ++c) {
            int idx = tid + c * 256;
            int ci = idx >> 3, c8 = (idx & 7) * 8;
            *(v8bf*)(out + ((size_t)b * 128 + ci) * 4096 + nBase + c8)
                = *(const v8bf*)(sOut + ci * 64 + c8);
        }
    }
}

// ---------------------------------------------------------------------------
// 2x2 maxpool kernels
// ---------------------------------------------------------------------------
__global__ __launch_bounds__(256)
void pool_n_kernel(const bf16_t* __restrict__ in, bf16_t* __restrict__ out) {
    size_t i = (size_t)blockIdx.x * 256 + threadIdx.x;   // (B,1024,128)
    if (i >= (size_t)16 * 1024 * 128) return;
    int ci = (int)(i & 127);
    int m  = (int)((i >> 7) & 1023);
    int b  = (int)(i >> 17);
    int ph = m >> 5, pw = m & 31;
    float mx = -3.0e38f;
#pragma unroll
    for (int dh = 0; dh < 2; ++dh)
#pragma unroll
        for (int dw = 0; dw < 2; ++dw) {
            int n = (2 * ph + dh) * 64 + (2 * pw + dw);
            mx = fmaxf(mx, (float)in[((size_t)b * 4096 + n) * 128 + ci]);
        }
    out[i] = (bf16_t)mx;
}

__global__ __launch_bounds__(256)
void pool_c_kernel(const bf16_t* __restrict__ in, bf16_t* __restrict__ out) {
    size_t i = (size_t)blockIdx.x * 256 + threadIdx.x;   // (B,128,1024)
    if (i >= (size_t)16 * 128 * 1024) return;
    int m  = (int)(i & 1023);
    int ci = (int)((i >> 10) & 127);
    int b  = (int)(i >> 17);
    int ph = m >> 5, pw = m & 31;
    float mx = -3.0e38f;
#pragma unroll
    for (int dh = 0; dh < 2; ++dh)
#pragma unroll
        for (int dw = 0; dw < 2; ++dw) {
            int n = (2 * ph + dh) * 64 + (2 * pw + dw);
            mx = fmaxf(mx, (float)in[((size_t)b * 128 + ci) * 4096 + n]);
        }
    out[i] = (bf16_t)mx;
}

// ---------------------------------------------------------------------------
// Kernel 3: flash attention.  Block = 1 batch x 64 query rows, 4 waves.
// Key/value panels of 32 streamed via async global->LDS double buffer
// (ASYNCcnt-paced); online softmax; z staged in LDS, stored ci-major.
// LDS: sQ 16K + sPG 32K + sE 4K = 52 KB.
// ---------------------------------------------------------------------------
__global__ __launch_bounds__(128)
void attn_kernel(const bf16_t* __restrict__ theta, // (B,4096,128)
                 const bf16_t* __restrict__ phi,   // (B,1024,128)
                 const bf16_t* __restrict__ g,     // (B,128,1024)
                 bf16_t* __restrict__ z) {         // (B,128,4096)
    const int NQ = 4096, M = 1024, MB = 32, NB = M / MB;
    __shared__ __align__(16) bf16_t sQ[64 * 128];   // [n][ci]
    // per buffer: [0..4095] = P panel [m][ci], [4096..8191] = G panel [ci][m]
    __shared__ __align__(16) bf16_t sPG[2][8192];
    __shared__ __align__(16) bf16_t sE[4][16 * 32]; // per-wave P stage [n][m]

    const int b  = blockIdx.y;
    const int rb = blockIdx.x;
    const int tid = threadIdx.x, wave = tid >> 5, lane = tid & 31;
    const int lncol = lane & 15, rowh = (lane >> 4) * 8;

    const bf16_t* qbase = theta + ((size_t)b * NQ + rb * 64) * 128;
    const bf16_t* pbase = phi + (size_t)b * M * 128;
    const bf16_t* gbase = g + (size_t)b * 128 * M;

#if USE_ASYNC_LDS
    // 8 async b128 instructions per thread per panel (ASYNCcnt += 8 per wave)
    auto issuePanel = [&](int cb, int buf) {
#pragma unroll
        for (int c = 0; c < 4; ++c) {              // P: 32x128 contiguous
            int idx = tid + c * 128;
            async_ld16(&sPG[buf][idx * 8],
                       pbase + (size_t)cb * MB * 128 + idx * 8);
        }
#pragma unroll
        for (int c = 0; c < 4; ++c) {              // G: 128 rows of 32
            int idx = tid + c * 128;
            int ci = idx >> 2, m8 = (idx & 3) * 8;
            async_ld16(&sPG[buf][4096 + ci * 32 + m8],
                       gbase + (size_t)ci * M + cb * MB + m8);
        }
    };
#else
    v8bf rp[4], rg[4];
    auto loadPG = [&](int cb) {
#pragma unroll
        for (int c = 0; c < 4; ++c) {
            int idx = tid + c * 128;
            rp[c] = *(const v8bf*)(pbase + (size_t)cb * MB * 128 + idx * 8);
            int ci = idx >> 2, m8 = (idx & 3) * 8;
            rg[c] = *(const v8bf*)(gbase + (size_t)ci * M + cb * MB + m8);
        }
    };
    auto storePG = [&](int buf) {
#pragma unroll
        for (int c = 0; c < 4; ++c) {
            int idx = tid + c * 128;
            *(v8bf*)&sPG[buf][idx * 8] = rp[c];
            int ci = idx >> 2, m8 = (idx & 3) * 8;
            *(v8bf*)&sPG[buf][4096 + ci * 32 + m8] = rg[c];
        }
    };
#endif

    // Q tile: contiguous 16KB region, straight b128 copy.
#pragma unroll
    for (int c = 0; c < 8; ++c) {
        int idx = tid + c * 128;
        *(v8bf*)&sQ[idx * 8] = *(const v8bf*)(qbase + idx * 8);
    }

    v8f O[8] = {};
    float rmax[8], rsum[8];
#pragma unroll
    for (int e = 0; e < 8; ++e) { rmax[e] = -3.0e38f; rsum[e] = 0.0f; }

#if USE_ASYNC_LDS
    issuePanel(0, 0);
#else
    loadPG(0);
    storePG(0);
    __syncthreads();
#endif

    for (int cb = 0; cb < NB; ++cb) {
        const int buf = cb & 1;
#if USE_ASYNC_LDS
        if (cb + 1 < NB) {
            issuePanel(cb + 1, buf ^ 1);           // fly under the WMMAs
            asm volatile("s_wait_asynccnt 0x8" ::: "memory"); // prev panel done
        } else {
            asm volatile("s_wait_asynccnt 0x0" ::: "memory");
        }
        __syncthreads();
#else
        if (cb + 1 < NB) loadPG(cb + 1);
#endif
        const bf16_t* sP = sPG[buf];
        const bf16_t* sG = sPG[buf] + 4096;

        // --- S = Q(16x128) @ P^T : 2 m-tiles x 4 K-steps ---
        v8f S[2] = {};
#pragma unroll
        for (int ks = 0; ks < 4; ++ks) {
            v16bf a = load_frag(sQ, wave * 16, ks * 32, 128);
#pragma unroll
            for (int mt = 0; mt < 2; ++mt) {
                v16bf bb = load_frag(sP, mt * 16, ks * 32, 128);
                S[mt] = wmma_bf16(a, bb, S[mt]);
            }
        }

        // --- online softmax ---
#pragma unroll
        for (int e = 0; e < 8; ++e) {
            float mx = fmaxf(S[0][e], S[1][e]);
#pragma unroll
            for (int d = 1; d < 16; d <<= 1) mx = fmaxf(mx, __shfl_xor(mx, d, 32));
            float mnew = fmaxf(rmax[e], mx);
            float corr = __expf(rmax[e] - mnew);
            rmax[e] = mnew;
            float s = 0.0f;
#pragma unroll
            for (int mt = 0; mt < 2; ++mt) {
                float p = __expf(S[mt][e] - mnew);
                S[mt][e] = p;
                s += p;
            }
#pragma unroll
            for (int d = 1; d < 16; d <<= 1) s += __shfl_xor(s, d, 32);
            rsum[e] = rsum[e] * corr + s;
#pragma unroll
            for (int ct = 0; ct < 8; ++ct) O[ct][e] *= corr;
        }

        // --- transpose P (D-layout -> A-layout) via per-wave LDS stage ---
        bf16_t* pe = &sE[wave][0];
#pragma unroll
        for (int mt = 0; mt < 2; ++mt)
#pragma unroll
            for (int e = 0; e < 8; ++e)
                pe[(e + rowh) * 32 + mt * 16 + lncol] = (bf16_t)S[mt][e];
        // wave-local LDS RAW: in-order within wave, no block barrier needed

        // --- O += P(16x32) @ G(32x128) ---
        {
            v16bf a = load_frag(pe, 0, 0, 32);
#pragma unroll
            for (int ct = 0; ct < 8; ++ct) {
                v16bf bb = load_frag(sG, ct * 16, 0, 32);
                O[ct] = wmma_bf16(a, bb, O[ct]);
            }
        }
        __syncthreads();                 // all waves done reading this buffer
#if !USE_ASYNC_LDS
        if (cb + 1 < NB) { storePG(buf ^ 1); __syncthreads(); }
#endif
    }

    // --- normalize, stage z tile [ci][n] in sPG[0] (8192 = 128x64), store ---
    bf16_t* sZ = sPG[0];
#pragma unroll
    for (int e = 0; e < 8; ++e) {
        float inv = 1.0f / rsum[e];
        int nl = wave * 16 + e + rowh;
#pragma unroll
        for (int ct = 0; ct < 8; ++ct) {
            int ci = ct * 16 + lncol;
            sZ[ci * 64 + nl] = (bf16_t)(O[ct][e] * inv);
        }
    }
    __syncthreads();
#pragma unroll
    for (int c = 0; c < 8; ++c) {
        int idx = tid + c * 128;
        int ci = idx >> 3, n8 = (idx & 7) * 8;
        *(v8bf*)(z + ((size_t)b * 128 + ci) * 4096 + rb * 64 + n8)
            = *(const v8bf*)(sZ + ci * 64 + n8);
    }
}

// ---------------------------------------------------------------------------
// Kernel 4/6: wz = W(256x128) @ z(128x4096) + Wb, per batch; tile 128o x 64n.
//   mode 0: per-channel sum/sumsq (LDS reduce + global atomics)
//   mode 1: out = wz*scale + shift + x  (fp32)
// ---------------------------------------------------------------------------
__global__ __launch_bounds__(256)
void wz_kernel(const bf16_t* __restrict__ zin,   // (B,128,4096)
               const float* __restrict__ Ww,     // (256,128)
               const float* __restrict__ Wb,     // (256)
               const float* __restrict__ xin,    // (B,256,4096)
               const float* __restrict__ scale,
               const float* __restrict__ shift,
               float* __restrict__ sums,         // [0..255]=sum [256..511]=sq
               float* __restrict__ outp,
               int mode) {
    const int K = 128, N = 4096;
    __shared__ __align__(16) bf16_t sA[2][128 * 32];
    __shared__ __align__(16) bf16_t sB[2][64 * 32];
    __shared__ float redS[128], redQ[128];

    const int b     = blockIdx.z;
    const int oBase = blockIdx.y * 128;
    const int nBase = blockIdx.x * 64;
    const int tid = threadIdx.x, wave = tid >> 5, lane = tid & 31;
    const int lncol = lane & 15, rowh = (lane >> 4) * 8;

    const int am = tid >> 1, ah = (tid & 1) * 16;
    const int kp = tid >> 4, n4 = (tid & 15) * 4;

    const float*  Arow = Ww + (size_t)(oBase + am) * K + ah;
    const bf16_t* Bcol = zin + (size_t)b * K * N + nBase + n4;

    v4f ra[4]; v4bf zb0, zb1;
    auto loadStage = [&](int k0) {
#pragma unroll
        for (int q = 0; q < 4; ++q) ra[q] = *(const v4f*)(Arow + k0 + q * 4);
        zb0 = *(const v4bf*)(Bcol + (size_t)(k0 + 2 * kp) * N);
        zb1 = *(const v4bf*)(Bcol + (size_t)(k0 + 2 * kp + 1) * N);
    };
    auto storeStage = [&](int buf) {
        v8bf pa0, pa1;
#pragma unroll
        for (int j = 0; j < 8; ++j) pa0[j] = (bf16_t)ra[j >> 2][j & 3];
#pragma unroll
        for (int j = 0; j < 8; ++j) pa1[j] = (bf16_t)ra[2 + (j >> 2)][j & 3];
        *(v8bf*)&sA[buf][am * 32 + ah]     = pa0;
        *(v8bf*)&sA[buf][am * 32 + ah + 8] = pa1;
#pragma unroll
        for (int j = 0; j < 4; ++j) {
            v2bf pk; pk[0] = zb0[j]; pk[1] = zb1[j];
            *(v2bf*)&sB[buf][(n4 + j) * 32 + 2 * kp] = pk;
        }
    };

    v8f acc[4] = {};
    loadStage(0);
    storeStage(0);
    __syncthreads();
    for (int s = 0; s < 4; ++s) {
        if (s < 3) loadStage((s + 1) * 32);
        v16bf a = load_frag(sA[s & 1], wave * 16, 0, 32);
#pragma unroll
        for (int nt = 0; nt < 4; ++nt) {
            v16bf bb = load_frag(sB[s & 1], nt * 16, 0, 32);
            acc[nt] = wmma_bf16(a, bb, acc[nt]);
        }
        if (s < 3) storeStage((s + 1) & 1);
        __syncthreads();
    }

#pragma unroll
    for (int r = 0; r < 8; ++r) {
        float bv = Wb[oBase + wave * 16 + r + rowh];
#pragma unroll
        for (int nt = 0; nt < 4; ++nt) acc[nt][r] += bv;
    }

    if (mode == 0) {
        if (tid < 128) { redS[tid] = 0.0f; redQ[tid] = 0.0f; }
        __syncthreads();
#pragma unroll
        for (int r = 0; r < 8; ++r) {
            int ol = wave * 16 + r + rowh;
            float sv = 0.0f, qv = 0.0f;
#pragma unroll
            for (int nt = 0; nt < 4; ++nt) {
                float v = acc[nt][r];
                sv += v; qv += v * v;
            }
            atomicAdd(&redS[ol], sv);
            atomicAdd(&redQ[ol], qv);
        }
        __syncthreads();
        if (tid < 128) {
            atomicAdd(&sums[oBase + tid],       redS[tid]);
            atomicAdd(&sums[256 + oBase + tid], redQ[tid]);
        }
    } else {
#pragma unroll
        for (int r = 0; r < 8; ++r) {
            int o = oBase + wave * 16 + r + rowh;
            float sc = scale[o], sh = shift[o];
            size_t base = ((size_t)b * 256 + o) * 4096;
#pragma unroll
            for (int nt = 0; nt < 4; ++nt) {
                int n = nBase + nt * 16 + lncol;
                outp[base + n] = acc[nt][r] * sc + sh + xin[base + n];
            }
        }
    }
}

// ---------------------------------------------------------------------------
__global__ void bn_prep_kernel(const float* __restrict__ sums,
                               const float* __restrict__ gamma,
                               const float* __restrict__ beta,
                               float* __restrict__ scale,
                               float* __restrict__ shift) {
    int o = blockIdx.x * blockDim.x + threadIdx.x;
    if (o < 256) {
        const float invN = 1.0f / (16.0f * 4096.0f);
        float mean = sums[o] * invN;
        float var  = sums[256 + o] * invN - mean * mean;
        float r    = rsqrtf(var + 1e-5f);
        float sc   = gamma[o] * r;
        scale[o] = sc;
        shift[o] = beta[o] - mean * sc;
    }
}

// ---------------------------------------------------------------------------
extern "C" void kernel_launch(void* const* d_in, const int* in_sizes, int n_in,
                              void* d_out, int out_size, void* d_ws, size_t ws_size,
                              hipStream_t stream) {
    const float* x       = (const float*)d_in[0];
    const float* y       = (const float*)d_in[1];
    const float* theta_w = (const float*)d_in[2];
    const float* theta_b = (const float*)d_in[3];
    const float* phi_w   = (const float*)d_in[4];
    const float* phi_b   = (const float*)d_in[5];
    const float* g_w     = (const float*)d_in[6];
    const float* g_b     = (const float*)d_in[7];
    const float* W_w     = (const float*)d_in[8];
    const float* W_b     = (const float*)d_in[9];
    const float* gamma   = (const float*)d_in[10];
    const float* beta    = (const float*)d_in[11];
    float* outp = (float*)d_out;

    char* ws = (char*)d_ws;
    size_t off = 0;
    auto carve = [&](size_t bytes) -> char* {
        char* p = ws + off;
        off += (bytes + 255) & ~(size_t)255;
        return p;
    };
    const size_t FULL = (size_t)16 * 4096 * 128 * sizeof(bf16_t); // 16 MB
    const size_t POOL = (size_t)16 * 1024 * 128 * sizeof(bf16_t); //  4 MB
    bf16_t* thetaB = (bf16_t*)carve(FULL);   // (B,4096,128)
    bf16_t* phiF   = (bf16_t*)carve(FULL);   // (B,4096,128)
    bf16_t* gF     = (bf16_t*)carve(FULL);   // (B,128,4096)
    bf16_t* phiP   = (bf16_t*)carve(POOL);   // (B,1024,128)
    bf16_t* gP     = (bf16_t*)carve(POOL);   // (B,128,1024)
    bf16_t* zB     = (bf16_t*)carve(FULL);   // (B,128,4096)
    float*  sums   = (float*)carve(512 * sizeof(float));
    float*  scale  = (float*)carve(256 * sizeof(float));
    float*  shift  = (float*)carve(256 * sizeof(float));
    (void)in_sizes; (void)n_in; (void)out_size; (void)ws_size;

    dim3 projGrid(64, 1, 16), projBlk(256);
    proj_kernel<<<projGrid, projBlk, 0, stream>>>(x, theta_w, theta_b, thetaB, 0);
    proj_kernel<<<projGrid, projBlk, 0, stream>>>(y, phi_w,   phi_b,   phiF,   0);
    proj_kernel<<<projGrid, projBlk, 0, stream>>>(y, g_w,     g_b,     gF,     1);

    const int poolElems = 16 * 1024 * 128;
    pool_n_kernel<<<(poolElems + 255) / 256, 256, 0, stream>>>(phiF, phiP);
    pool_c_kernel<<<(poolElems + 255) / 256, 256, 0, stream>>>(gF, gP);

    hipMemsetAsync(sums, 0, 512 * sizeof(float), stream);

    attn_kernel<<<dim3(64, 16), 128, 0, stream>>>(thetaB, phiP, gP, zB);

    dim3 wzGrid(64, 2, 16), wzBlk(256);
    wz_kernel<<<wzGrid, wzBlk, 0, stream>>>(zB, W_w, W_b, x, scale, shift,
                                            sums, outp, 0);
    bn_prep_kernel<<<1, 256, 0, stream>>>(sums, gamma, beta, scale, shift);
    wz_kernel<<<wzGrid, wzBlk, 0, stream>>>(zB, W_w, W_b, x, scale, shift,
                                            sums, outp, 1);
}
